// Bern_AllHist_GLM_41034117546317
// MI455X (gfx1250) — matrix operand order repaired and checked by reading, not compiled, not run on hardware
//
#include <hip/hip_runtime.h>
#include <hip/hip_bf16.h>

// ---------------------------------------------------------------------------
// Bern_AllHist_GLM on MI455X (gfx1250).
//
// P[t,s] = sigmoid( Theta[s] + sum_{j=0}^{199} X[t-1-j,:] @ G[:,s,j] )
// X = [syn_e(20) | syn_i(20) | Z(20) | 0(4)]  (64 bf16 channels)
//   G[c,s,j] = delta(c,s)*e_kern[s,j]                       (c in [0,20))
//            = delta(c-20,s)*i_kern[s,j]                    (c in [20,40))
//            = C_den[s,c-40]*spk_kern[c-40,j] + delta(c-40,s)*hist_kern[s,j]
// im2col bf16 WMMA GEMM; ~240 GFLOP bf16, ~580 MB HBM traffic (23.3 TB/s
// floor; X and the 800 KB weight tensor are L2-resident on 192 MB L2).
// Conv LDS staging: GLOBAL_LOAD_ASYNC_TO_LDS_B128 with DOUBLE-BUFFERED weight
// slabs so the async DMA overlaps the WMMA stream (ASYNCcnt/s_wait_asynccnt).
// ---------------------------------------------------------------------------

typedef __attribute__((ext_vector_type(16))) __bf16 v16bf;
typedef __attribute__((ext_vector_type(8)))  float  v8f;

union BF16x16 {
    unsigned short u[16];
    __bf16         h[16];
    uint4          q[2];
    v16bf          v;
};

__device__ __forceinline__ unsigned short f2bf(float f) {
    union { float f; unsigned u; } c; c.f = f;
    unsigned r = c.u + 0x7FFFu + ((c.u >> 16) & 1u);   // round-to-nearest-even
    return (unsigned short)(r >> 16);
}

__device__ __forceinline__ int imin(int a, int b) { return a < b ? a : b; }

// Async global->LDS copy of 16 bytes per lane (CDNA5 async path).
// VDST = LDS byte address, VADDR = 64-bit global address. Tracked by ASYNCcnt.
__device__ __forceinline__ void async_ld_b128(void* lds_ptr, const void* gptr) {
    unsigned lds_addr = (unsigned)(unsigned long long)(uintptr_t)lds_ptr;
    unsigned long long gaddr = (unsigned long long)(uintptr_t)gptr;
    asm volatile("global_load_async_to_lds_b128 %0, %1, off"
                 :: "v"(lds_addr), "v"(gaddr) : "memory");
}
__device__ __forceinline__ void wait_async0() {
    asm volatile("s_wait_asynccnt 0x0" ::: "memory");
}

#define SUB    20
#define KLEN   200           // T_no (fixed by harness setup)
#define NBAS   13
#define XCH    64            // padded channel count of X
#define XPAD   256           // zero halo rows before/after X
#define MBLK   64            // output rows per conv block
#define WIN    (KLEN + MBLK) // 264 LDS rows
#define JT     2             // taps per double-buffered LDS slab
#define SLAB   (JT * 2 * 2 * 512)   // bf16 elements per slab (4096)
#define PI_F   3.14159265358979323846f

// ---------------------------------------------------------------------------
// Kernel 1: build filters (-> d_out tail), bake conv weights G into the WMMA
// B-fragment layout, bake zero-padded Ce/Ci B-fragments, zero the X halo.
//   conv W layout : idx = ((j*2+cb)*2+nt)*512 + n*32 + k   c=cb*32+k, s=nt*16+n
//   Ce/Ci layout  : idx = (kb*2+nt)*512 + n*32 + k         kg=kb*32+k
// ---------------------------------------------------------------------------
__global__ __launch_bounds__(256) void setup_kernel(
    const float* __restrict__ C_den,  const float* __restrict__ Tau,
    const float* __restrict__ Delta,  const float* __restrict__ Wsyn,
    const float* __restrict__ Wspk,   const float* __restrict__ Whist,
    const float* __restrict__ Ce,     const float* __restrict__ Ci,
    float* __restrict__ out_filters,  unsigned short* __restrict__ Wg,
    unsigned short* __restrict__ CeB, unsigned short* __restrict__ CiB,
    unsigned short* __restrict__ Xg,  int T, int En, int In, int kbE, int kbI)
{
    __shared__ float ek[SUB * KLEN];
    __shared__ float ik[SUB * KLEN];
    __shared__ float sk[SUB * KLEN];
    __shared__ float hk[SUB * KLEN];

    const int tid = threadIdx.x;

    for (int idx = tid; idx < SUB * KLEN; idx += 256) {
        int s = idx / KLEN, j = idx % KLEN;
        float t    = (float)j;
        float te   = fmaxf(t - Delta[s * 2 + 0], 0.f);
        float ti   = fmaxf(t - Delta[s * 2 + 1], 0.f);
        float taue = Tau[s * 2 + 0], taui = Tau[s * 2 + 1];
        float tte  = te / (taue * taue);
        float tti  = ti / (taui * taui);
        float we   = Wsyn[s * 2 + 0], wi = Wsyn[s * 2 + 1];
        float ev   =  tte * __expf(-tte) * we * we;
        float iv   = -(tti * __expf(-tti) * wi * wi);

        float raw = 3.0f * __logf(t + 1.0f);
        float sv = 0.f, hv = 0.f;
        #pragma unroll
        for (int b = 0; b < NBAS; ++b) {
            float phi = 0.5f * PI_F * (float)b;
            float d   = raw - phi;
            float bas = (d >= -PI_F && d <= PI_F) ? (0.5f * __cosf(d) + 0.5f) : 0.f;
            sv += Wspk[s * NBAS + b] * bas;
            hv += Whist[s * NBAS + b] * bas;
        }
        ek[idx] = ev; ik[idx] = iv; sk[idx] = sv; hk[idx] = hv;
        out_filters[idx]                  = ev;   // rows  0..19 : e_kern
        out_filters[SUB * KLEN + idx]     = iv;   // rows 20..39 : i_kern
        out_filters[2 * SUB * KLEN + idx] = sv;   // rows 40..59 : spk_kern
        out_filters[3 * SUB * KLEN + idx] = hv;   // rows 60..79 : hist_kern
    }

    // zero X halo (head XPAD rows + tail XPAD rows)
    {
        uint4* head = (uint4*)Xg;
        uint4* tail = (uint4*)(Xg + (size_t)(XPAD + T) * XCH);
        uint4  z    = make_uint4(0u, 0u, 0u, 0u);
        const int nq = XPAD * XCH * 2 / 16;   // 2048 uint4
        for (int i = tid; i < nq; i += 256) { head[i] = z; tail[i] = z; }
    }

    // projection B-fragments (zero-padded): kills clamped-K garbage in A
    for (int idx = tid; idx < kbE * 1024; idx += 256) {
        int k  = idx & 31;
        int n  = (idx >> 5) & 15;
        int nt = (idx >> 9) & 1;
        int kb = idx >> 10;
        int s  = nt * 16 + n;
        int kg = kb * 32 + k;
        CeB[idx] = (s < SUB && kg < En) ? f2bf(Ce[s * En + kg]) : (unsigned short)0;
    }
    for (int idx = tid; idx < kbI * 1024; idx += 256) {
        int k  = idx & 31;
        int n  = (idx >> 5) & 15;
        int nt = (idx >> 9) & 1;
        int kb = idx >> 10;
        int s  = nt * 16 + n;
        int kg = kb * 32 + k;
        CiB[idx] = (s < SUB && kg < In) ? f2bf(Ci[s * In + kg]) : (unsigned short)0;
    }
    __syncthreads();

    // conv weight tensor in B-fragment layout
    for (int idx = tid; idx < KLEN * 2 * 2 * 512; idx += 256) {
        int k  = idx & 31;
        int n  = (idx >> 5) & 15;
        int nt = (idx >> 9) & 1;
        int cb = (idx >> 10) & 1;
        int j  = idx >> 11;
        int c  = cb * 32 + k;
        int s  = nt * 16 + n;
        float g = 0.f;
        if (s < SUB) {
            if (c < SUB) {
                if (c == s) g = ek[s * KLEN + j];
            } else if (c < 2 * SUB) {
                if (c - SUB == s) g = ik[s * KLEN + j];
            } else if (c < 3 * SUB) {
                int cz = c - 2 * SUB;
                g = C_den[s * SUB + cz] * sk[cz * KLEN + j];
                if (cz == s) g += hk[s * KLEN + j];
            }
        }
        Wg[idx] = f2bf(g);
    }
}

// ---------------------------------------------------------------------------
// Kernel 2: projections via WMMA.  One wave = one 16-row time tile.
// A loads are UNCONDITIONAL with clamped addresses (no EXEC divergence);
// out-of-range K columns are annihilated by the zero-padded B fragments,
// out-of-range rows only affect D rows that are never stored.
// ---------------------------------------------------------------------------
__global__ __launch_bounds__(256) void proj_kernel(
    const float* __restrict__ Se, const float* __restrict__ Si,
    const float* __restrict__ Z,
    const unsigned short* __restrict__ CeB,
    const unsigned short* __restrict__ CiB,
    unsigned short* __restrict__ Xg,
    int T, int En, int In, int kbE, int kbI)
{
    const int tid  = threadIdx.x;
    const int wave = tid >> 5;
    const int lane = tid & 31;
    const int half = lane >> 4;
    const int l16  = lane & 15;
    const int t0   = blockIdx.x * 128 + wave * 16;
    if (t0 >= T) return;   // uniform per wave; no __syncthreads in this kernel

    const int rowA = imin(t0 + l16, T - 1);     // clamped, always in-bounds

    v8f accE[2]; v8f accI[2];
    accE[0] = (v8f){}; accE[1] = (v8f){}; accI[0] = (v8f){}; accI[1] = (v8f){};

    // ---- excitatory projection ----
    {
        const float* sp = Se + (size_t)rowA * En;
        for (int kb = 0; kb < kbE; ++kb) {
            BF16x16 af;
            #pragma unroll
            for (int e = 0; e < 8; ++e) {
                int k  = kb * 32 + half * 8 + e;
                af.h[e]     = (__bf16)sp[imin(k,      En - 1)];
                af.h[8 + e] = (__bf16)sp[imin(k + 16, En - 1)];
            }
            #pragma unroll
            for (int nt = 0; nt < 2; ++nt) {
                BF16x16 bf;
                const unsigned short* bs =
                    CeB + (kb * 2 + nt) * 512 + l16 * 32 + half * 8;
                bf.q[0] = *(const uint4*)(bs);
                bf.q[1] = *(const uint4*)(bs + 16);
                accE[nt] = __builtin_amdgcn_wmma_f32_16x16x32_bf16(
                    false, af.v, false, bf.v, (short)0, accE[nt], false, false);
            }
        }
    }

    // ---- inhibitory projection ----
    {
        const float* sp = Si + (size_t)rowA * In;
        for (int kb = 0; kb < kbI; ++kb) {
            BF16x16 af;
            #pragma unroll
            for (int e = 0; e < 8; ++e) {
                int k  = kb * 32 + half * 8 + e;
                af.h[e]     = (__bf16)sp[imin(k,      In - 1)];
                af.h[8 + e] = (__bf16)sp[imin(k + 16, In - 1)];
            }
            #pragma unroll
            for (int nt = 0; nt < 2; ++nt) {
                BF16x16 bf;
                const unsigned short* bs =
                    CiB + (kb * 2 + nt) * 512 + l16 * 32 + half * 8;
                bf.q[0] = *(const uint4*)(bs);
                bf.q[1] = *(const uint4*)(bs + 16);
                accI[nt] = __builtin_amdgcn_wmma_f32_16x16x32_bf16(
                    false, af.v, false, bf.v, (short)0, accI[nt], false, false);
            }
        }
    }

    // ---- store D tiles into X (bf16).  D layout: lane n=l16, VGPR v -> M=8*half+v.
    __bf16* Xh = (__bf16*)Xg;
    #pragma unroll
    for (int nt = 0; nt < 2; ++nt) {
        int c = nt * 16 + l16;
        if (c < SUB) {
            #pragma unroll
            for (int v = 0; v < 8; ++v) {
                int row = t0 + half * 8 + v;
                if (row < T) {
                    size_t base = (size_t)(XPAD + row) * XCH;
                    Xh[base + c]       = (__bf16)accE[nt][v];
                    Xh[base + SUB + c] = (__bf16)accI[nt][v];
                }
            }
        }
    }

    // ---- Z copy + zero channels 60..63 ----
    for (int i = lane; i < 16 * 24; i += 32) {
        int r = i / 24, c = i % 24;
        int row = t0 + r;
        if (row < T) {
            float z = (c < SUB) ? Z[(size_t)row * SUB + c] : 0.f;
            Xh[(size_t)(XPAD + row) * XCH + 2 * SUB + c] = (__bf16)z;
        }
    }
}

// ---------------------------------------------------------------------------
// Kernel 3: fused causal-conv GEMM + Theta + sigmoid.
// Block = 256 threads (8 waves) = 4 M-tiles x 2 N-tiles over MBLK=64 rows.
// X window + DOUBLE-BUFFERED weight slabs staged with async b128 loads, so
// the next slab's DMA overlaps the current slab's WMMAs.
// LDS: 33792 (X) + 2*8192 (W slabs) = 50176 B.
// ---------------------------------------------------------------------------
__global__ __launch_bounds__(256) void conv_kernel(
    const unsigned short* __restrict__ Xg, const unsigned short* __restrict__ Wg,
    const float* __restrict__ Theta, float* __restrict__ P, int T)
{
    __shared__ unsigned short Xs[WIN * XCH];
    __shared__ unsigned short Ws[2][SLAB];

    const int tid  = threadIdx.x;
    const int wave = tid >> 5;
    const int lane = tid & 31;
    const int half = lane >> 4;
    const int l16  = lane & 15;
    const int t0   = blockIdx.x * MBLK;
    const int mt   = wave >> 1;   // 0..3  (16-row sub-tile)
    const int nt   = wave & 1;    // 0..1  (16-col sub-tile)

    // async load of the X window: global rows [XPAD+t0-KLEN, XPAD+t0+MBLK)
    {
        const unsigned short* src = Xg + (size_t)(t0 + XPAD - KLEN) * XCH;
        const int nq = WIN * XCH / 8;   // 2112 x 16B chunks
        for (int i = tid; i < nq; i += 256)
            async_ld_b128(&Xs[i * 8], src + (size_t)i * 8);
    }
    // async stage of slab 0
    for (int i = tid; i < SLAB / 8; i += 256)
        async_ld_b128(&Ws[0][i * 8], Wg + (size_t)i * 8);

    v8f acc = (v8f){};

    for (int jb = 0; jb < KLEN / JT; ++jb) {
        wait_async0();        // current slab (and X window on jb==0) resident
        __syncthreads();      // all waves past previous compute + writes visible

        if (jb + 1 < KLEN / JT) {   // prefetch next slab into the other buffer
            const unsigned short* src = Wg + (size_t)(jb + 1) * SLAB;
            for (int i = tid; i < SLAB / 8; i += 256)
                async_ld_b128(&Ws[(jb + 1) & 1][i * 8], src + (size_t)i * 8);
        }

        const unsigned short* Wsb = Ws[jb & 1];
        #pragma unroll
        for (int jj = 0; jj < JT; ++jj) {
            const int j = jb * JT + jj;
            #pragma unroll
            for (int cb = 0; cb < 2; ++cb) {
                // B fragment: lane column n=l16, k at half*8 and 16+half*8 (8 each)
                BF16x16 bf;
                const unsigned short* wsrc =
                    Wsb + ((jj * 2 + cb) * 2 + nt) * 512 + l16 * 32 + half * 8;
                bf.q[0] = *(const uint4*)(wsrc);
                bf.q[1] = *(const uint4*)(wsrc + 16);
                // A fragment: out row t0+mt*16+l16 needs X[t-1-j]
                BF16x16 af;
                const int ldsrow = mt * 16 + l16 - 1 - j + KLEN;   // in [0, WIN)
                const unsigned short* xsrc =
                    Xs + ldsrow * XCH + cb * 32 + half * 8;
                af.q[0] = *(const uint4*)(xsrc);
                af.q[1] = *(const uint4*)(xsrc + 16);

                acc = __builtin_amdgcn_wmma_f32_16x16x32_bf16(
                    false, af.v, false, bf.v, (short)0, acc, false, false);
            }
        }
    }

    // epilogue: + Theta, sigmoid, store (D: lane n=l16, VGPR v -> M=8*half+v)
    const int s = nt * 16 + l16;
    if (s < SUB) {
        const float th = Theta[s];
        #pragma unroll
        for (int v = 0; v < 8; ++v) {
            int row = t0 + mt * 16 + half * 8 + v;
            if (row < T) {
                float x = acc[v] + th;
                P[(size_t)row * SUB + s] = 1.f / (1.f + __expf(-x));
            }
        }
    }
}

// ---------------------------------------------------------------------------
extern "C" void kernel_launch(void* const* d_in, const int* in_sizes, int n_in,
                              void* d_out, int out_size, void* d_ws, size_t ws_size,
                              hipStream_t stream)
{
    const float* Se    = (const float*)d_in[0];
    const float* Si    = (const float*)d_in[1];
    const float* Z     = (const float*)d_in[2];
    const float* Cden  = (const float*)d_in[3];
    const float* Ce    = (const float*)d_in[4];
    const float* Ci    = (const float*)d_in[5];
    const float* Tau   = (const float*)d_in[6];
    const float* Delta = (const float*)d_in[7];
    const float* Wsyn  = (const float*)d_in[8];
    const float* Wspk  = (const float*)d_in[9];
    const float* Whist = (const float*)d_in[10];
    const float* Theta = (const float*)d_in[11];

    const int T   = in_sizes[2] / SUB;      // Z is (T, 20)
    const int En  = in_sizes[0] / T;        // 200
    const int In  = in_sizes[1] / T;        // 50
    const int kbE = (En + 31) / 32;
    const int kbI = (In + 31) / 32;

    float* P           = (float*)d_out;
    float* out_filters = P + (size_t)T * SUB;

    // workspace: X (bf16) | conv weights | CeB | CiB
    unsigned short* Xg = (unsigned short*)d_ws;
    size_t xElems = (size_t)(T + 2 * XPAD) * XCH;
    xElems = (xElems + 7) & ~(size_t)7;     // keep 16B alignment
    unsigned short* Wg  = Xg + xElems;
    unsigned short* CeB = Wg + (size_t)KLEN * 2 * 2 * 512;
    unsigned short* CiB = CeB + (size_t)kbE * 1024;

    setup_kernel<<<1, 256, 0, stream>>>(Cden, Tau, Delta, Wsyn, Wspk, Whist,
                                        Ce, Ci, out_filters, Wg, CeB, CiB,
                                        Xg, T, En, In, kbE, kbI);
    proj_kernel<<<(T + 127) / 128, 256, 0, stream>>>(Se, Si, Z, CeB, CiB, Xg,
                                                     T, En, In, kbE, kbI);
    conv_kernel<<<(T + MBLK - 1) / MBLK, 256, 0, stream>>>(Xg, Wg, Theta, P, T);
}
